// NodeModelLtp_21655225106535
// MI455X (gfx1250) — matrix operand order repaired and checked by reading, not compiled, not run on hardware
//
#include <hip/hip_runtime.h>

typedef __bf16 bf16_t;
typedef __attribute__((ext_vector_type(16))) bf16_t v16bf;
typedef __attribute__((ext_vector_type(8)))  bf16_t v8bf;
typedef __attribute__((ext_vector_type(8)))  float  v8f;

#define NN 50000
#define NE 800000
#define FD 128
#define HD 128
#define LN_EPS 1e-5f

// Convert 8 consecutive floats (16B-aligned) to bf16, with optional scale.
__device__ __forceinline__ v8bf cvt8(const float* p, float s) {
  float4 a = *(const float4*)p;
  float4 b = *(const float4*)(p + 4);
  v8bf r;
  r[0] = (bf16_t)(a.x * s); r[1] = (bf16_t)(a.y * s);
  r[2] = (bf16_t)(a.z * s); r[3] = (bf16_t)(a.w * s);
  r[4] = (bf16_t)(b.x * s); r[5] = (bf16_t)(b.y * s);
  r[6] = (bf16_t)(b.z * s); r[7] = (bf16_t)(b.w * s);
  return r;
}

__device__ __forceinline__ v16bf cat16(v8bf lo, v8bf hi) {
  v16bf r;
#pragma unroll
  for (int i = 0; i < 8; ++i) { r[i] = lo[i]; r[i + 8] = hi[i]; }
  return r;
}

// Bias + per-row LayerNorm + atomic scatter-add of one 16x128 C tile.
// C layout: VGPR r, lane l -> M = r + 8*(l>=16), N = t*16 + (l&15).
__device__ __forceinline__ void ln_scatter_tile(
    v8f (&d)[8],
    const float* __restrict__ b2,
    const float* __restrict__ gam,
    const float* __restrict__ bet,
    const int*   __restrict__ rowIdx, int eFirst,
    float* __restrict__ summed, int m, int hlf)
{
#pragma unroll
  for (int t = 0; t < 8; ++t) {
    const float bias = b2[t * 16 + m];
#pragma unroll
    for (int r = 0; r < 8; ++r) d[t][r] += bias;
  }
#pragma unroll
  for (int r = 0; r < 8; ++r) {
    float s = 0.0f, ss = 0.0f;
#pragma unroll
    for (int t = 0; t < 8; ++t) { float v = d[t][r]; s += v; ss += v * v; }
#pragma unroll
    for (int off = 1; off < 16; off <<= 1) {   // row lives in one 16-lane half
      s  += __shfl_xor(s,  off, 32);
      ss += __shfl_xor(ss, off, 32);
    }
    const float mu   = s * (1.0f / HD);
    const float var  = ss * (1.0f / HD) - mu * mu;
    const float rstd = rsqrtf(var + LN_EPS);
    const long long dst = (long long)rowIdx[eFirst + r + 8 * hlf];
    float* drow = summed + dst * HD;
#pragma unroll
    for (int t = 0; t < 8; ++t) {
      const int n = t * 16 + m;
      const float v = (d[t][r] - mu) * rstd * gam[n] + bet[n];
      atomicAdd(drow + n, v);
    }
  }
}

// ---------------------------------------------------------------------------
// Edge kernel: one wave = 32 edges (two 16-row A tiles sharing every B frag,
// halving L2 weight traffic and doubling WMMA work per B load).
//   h  = relu([x[col] | edge_attr] @ W1a + b1a)            (K = 256)
//   y  = layernorm(h @ W2a + b2a)                          (K = 128)
//   atomicAdd into summed[row], counts[row]
// ---------------------------------------------------------------------------
__global__ __launch_bounds__(32)
void edge_kernel(const float* __restrict__ x,
                 const int*   __restrict__ eidx,   // [2, E] flat: row then col
                 const float* __restrict__ eattr,  // [E, 128]
                 const bf16_t* __restrict__ W1,    // [256, 128] bf16
                 const float* __restrict__ b1,
                 const bf16_t* __restrict__ W2,    // [128, 128] bf16
                 const float* __restrict__ b2,
                 const float* __restrict__ gam,
                 const float* __restrict__ bet,
                 float* __restrict__ summed,       // [N, 128]
                 float* __restrict__ counts)       // [N]
{
  __shared__ bf16_t hbuf[32 * HD];   // 32x128 bf16 staging tile (8KB)

  const int lane = threadIdx.x & 31;
  const int hlf  = lane >> 4;
  const int m    = lane & 15;
  const int kb   = hlf * 8;          // A-frag K sub-offset per half-wave

  const int eBase = blockIdx.x * 32;
  const int eA    = eBase + m;       // rows 0..15
  const int eB    = eBase + 16 + m;  // rows 16..31

  const int* rowIdx = eidx;          // destination nodes
  const int* colIdx = eidx + NE;     // source nodes

  const float* xrowA = x + (long long)colIdx[eA] * FD;
  const float* arowA = eattr + (long long)eA * HD;
  const float* xrowB = x + (long long)colIdx[eB] * FD;
  const float* arowB = eattr + (long long)eB * HD;

  // ---------------- GEMM 1: [32 x 256] @ [256 x 128] -----------------------
  v8f acc0[8] = {};
  v8f acc1[8] = {};
  for (int kc = 0; kc < 8; ++kc) {
    const int k0 = kc * 32;
    const int K0 = k0 + kb;
    const int K1 = k0 + 16 + kb;
    const float* pA0 = (K0 < FD) ? (xrowA + K0) : (arowA + (K0 - FD));
    const float* pA1 = (K1 < FD) ? (xrowA + K1) : (arowA + (K1 - FD));
    const float* pB0 = (K0 < FD) ? (xrowB + K0) : (arowB + (K0 - FD));
    const float* pB1 = (K1 < FD) ? (xrowB + K1) : (arowB + (K1 - FD));
    v16bf aA = cat16(cvt8(pA0, 1.0f), cvt8(pA1, 1.0f));
    v16bf aB = cat16(cvt8(pB0, 1.0f), cvt8(pB1, 1.0f));
    const bf16_t* wrow = W1 + (k0 + kb) * HD + m;   // B-frag: lane = col N=t*16+m, K table order
#pragma unroll
    for (int t = 0; t < 8; ++t) {
      v16bf bfrag;
      for (int s = 0; s < 16; ++s) bfrag[s] = wrow[((s & 7) + 16 * (s >> 3)) * HD + t * 16];
      acc0[t] = __builtin_amdgcn_wmma_f32_16x16x32_bf16(
          false, aA, false, bfrag, (short)0, acc0[t], false, false);
      acc1[t] = __builtin_amdgcn_wmma_f32_16x16x32_bf16(
          false, aB, false, bfrag, (short)0, acc1[t], false, false);
    }
  }

  // bias + relu, stage both tiles to LDS row-major (C-layout -> A-layout)
#pragma unroll
  for (int t = 0; t < 8; ++t) {
    const int n = t * 16 + m;
    const float bias = b1[n];
#pragma unroll
    for (int r = 0; r < 8; ++r) {
      float v0 = acc0[t][r] + bias;
      float v1 = acc1[t][r] + bias;
      v0 = v0 > 0.0f ? v0 : 0.0f;
      v1 = v1 > 0.0f ? v1 : 0.0f;
      hbuf[(r + 8 * hlf) * HD + n]        = (bf16_t)v0;
      hbuf[(16 + r + 8 * hlf) * HD + n]   = (bf16_t)v1;
    }
  }
  __syncthreads();

  // ---------------- GEMM 2: [32 x 128] @ [128 x 128] -----------------------
  v8f d0[8] = {};
  v8f d1[8] = {};
  for (int kc = 0; kc < 4; ++kc) {
    const int k0 = kc * 32;
    v16bf aA = cat16(*(const v8bf*)(hbuf + m * HD + k0 + kb),
                     *(const v8bf*)(hbuf + m * HD + k0 + 16 + kb));
    v16bf aB = cat16(*(const v8bf*)(hbuf + (16 + m) * HD + k0 + kb),
                     *(const v8bf*)(hbuf + (16 + m) * HD + k0 + 16 + kb));
    const bf16_t* wrow = W2 + (k0 + kb) * HD + m;
#pragma unroll
    for (int t = 0; t < 8; ++t) {
      v16bf bfrag;
      for (int s = 0; s < 16; ++s) bfrag[s] = wrow[((s & 7) + 16 * (s >> 3)) * HD + t * 16];
      d0[t] = __builtin_amdgcn_wmma_f32_16x16x32_bf16(
          false, aA, false, bfrag, (short)0, d0[t], false, false);
      d1[t] = __builtin_amdgcn_wmma_f32_16x16x32_bf16(
          false, aB, false, bfrag, (short)0, d1[t], false, false);
    }
  }

  // bias + LayerNorm + atomic scatter for both 16-row tiles
  ln_scatter_tile(d0, b2, gam, bet, rowIdx, eBase,      summed, m, hlf);
  ln_scatter_tile(d1, b2, gam, bet, rowIdx, eBase + 16, summed, m, hlf);

  // counts: one +1 per edge, 32 edges <-> 32 lanes
  atomicAdd(counts + rowIdx[eBase + lane], 1.0f);
}

// ---------------------------------------------------------------------------
// Node kernel: one wave = 16 nodes (50000/32 is not integral, so stay at 16
// to keep EXEC all-ones around WMMA; node path is only ~6% of total FLOPs).
//   z   = [x | summed/clamp(counts) | u]   (K = 384)
//   out = layernorm(relu(z @ W1b + b1b) @ W2b + b2b)
// ---------------------------------------------------------------------------
__global__ __launch_bounds__(32)
void node_kernel(const float* __restrict__ x,
                 const float* __restrict__ u,
                 const float* __restrict__ summed,
                 const float* __restrict__ counts,
                 const bf16_t* __restrict__ W1,    // [384, 128] bf16
                 const float* __restrict__ b1,
                 const bf16_t* __restrict__ W2,    // [128, 128] bf16
                 const float* __restrict__ b2,
                 const float* __restrict__ gam,
                 const float* __restrict__ bet,
                 float* __restrict__ out)          // [N, 128]
{
  __shared__ bf16_t hbuf[16 * HD];

  const int lane = threadIdx.x & 31;
  const int hlf  = lane >> 4;
  const int m    = lane & 15;
  const int kb   = hlf * 8;

  const int nBase = blockIdx.x * 16;
  const int node  = nBase + m;

  const float* xrow = x + (long long)node * FD;
  const float* srow = summed + (long long)node * HD;
  const float* urow = u + (long long)node * HD;
  const float inv = 1.0f / fmaxf(counts[node], 1.0f);   // scatter-mean divisor

  // ---------------- GEMM 1: [16 x 384] @ [384 x 128] -----------------------
  v8f acc[8] = {};
  for (int kc = 0; kc < 12; ++kc) {
    const int k0 = kc * 32;
    const int K0 = k0 + kb;
    const int K1 = k0 + 16 + kb;
    v8bf lo = (K0 < FD)     ? cvt8(xrow + K0, 1.0f)
            : (K0 < 2 * FD) ? cvt8(srow + (K0 - FD), inv)
                            : cvt8(urow + (K0 - 2 * FD), 1.0f);
    v8bf hi = (K1 < FD)     ? cvt8(xrow + K1, 1.0f)
            : (K1 < 2 * FD) ? cvt8(srow + (K1 - FD), inv)
                            : cvt8(urow + (K1 - 2 * FD), 1.0f);
    v16bf afrag = cat16(lo, hi);
    const bf16_t* wrow = W1 + (k0 + kb) * HD + m;
#pragma unroll
    for (int t = 0; t < 8; ++t) {
      v16bf bfrag;
      for (int s = 0; s < 16; ++s) bfrag[s] = wrow[((s & 7) + 16 * (s >> 3)) * HD + t * 16];
      acc[t] = __builtin_amdgcn_wmma_f32_16x16x32_bf16(
          false, afrag, false, bfrag, (short)0, acc[t], false, false);
    }
  }

#pragma unroll
  for (int t = 0; t < 8; ++t) {
    const int n = t * 16 + m;
    const float bias = b1[n];
#pragma unroll
    for (int r = 0; r < 8; ++r) {
      float v = acc[t][r] + bias;
      v = v > 0.0f ? v : 0.0f;
      hbuf[(r + 8 * hlf) * HD + n] = (bf16_t)v;
    }
  }
  __syncthreads();

  // ---------------- GEMM 2: [16 x 128] @ [128 x 128] -----------------------
  v8f d[8] = {};
  for (int kc = 0; kc < 4; ++kc) {
    const int k0 = kc * 32;
    v16bf afrag = cat16(*(const v8bf*)(hbuf + m * HD + k0 + kb),
                        *(const v8bf*)(hbuf + m * HD + k0 + 16 + kb));
    const bf16_t* wrow = W2 + (k0 + kb) * HD + m;
#pragma unroll
    for (int t = 0; t < 8; ++t) {
      v16bf bfrag;
      for (int s = 0; s < 16; ++s) bfrag[s] = wrow[((s & 7) + 16 * (s >> 3)) * HD + t * 16];
      d[t] = __builtin_amdgcn_wmma_f32_16x16x32_bf16(
          false, afrag, false, bfrag, (short)0, d[t], false, false);
    }
  }

#pragma unroll
  for (int t = 0; t < 8; ++t) {
    const float bias = b2[t * 16 + m];
#pragma unroll
    for (int r = 0; r < 8; ++r) d[t][r] += bias;
  }

#pragma unroll
  for (int r = 0; r < 8; ++r) {
    float s = 0.0f, ss = 0.0f;
#pragma unroll
    for (int t = 0; t < 8; ++t) { float v = d[t][r]; s += v; ss += v * v; }
#pragma unroll
    for (int off = 1; off < 16; off <<= 1) {
      s  += __shfl_xor(s,  off, 32);
      ss += __shfl_xor(ss, off, 32);
    }
    const float mu   = s * (1.0f / HD);
    const float var  = ss * (1.0f / HD) - mu * mu;
    const float rstd = rsqrtf(var + LN_EPS);
    float* orow = out + (long long)(nBase + r + 8 * hlf) * HD;
#pragma unroll
    for (int t = 0; t < 8; ++t) {
      const int n = t * 16 + m;
      orow[n] = (d[t][r] - mu) * rstd * gam[n] + bet[n];
    }
  }
}

// ---------------------------------------------------------------------------
// Prep kernels
// ---------------------------------------------------------------------------
__global__ void zero_f32(float* __restrict__ p, int n) {
  int i = blockIdx.x * blockDim.x + threadIdx.x;
  if (i < n) p[i] = 0.0f;
}

__global__ void cvt_f32_bf16(const float* __restrict__ s, bf16_t* __restrict__ d, int n) {
  int i = blockIdx.x * blockDim.x + threadIdx.x;
  if (i < n) d[i] = (bf16_t)s[i];
}

// ---------------------------------------------------------------------------
extern "C" void kernel_launch(void* const* d_in, const int* in_sizes, int n_in,
                              void* d_out, int out_size, void* d_ws, size_t ws_size,
                              hipStream_t stream) {
  (void)in_sizes; (void)n_in; (void)out_size; (void)ws_size;

  const float* x    = (const float*)d_in[0];
  const int*   eidx = (const int*)  d_in[1];
  const float* eatt = (const float*)d_in[2];
  const float* u    = (const float*)d_in[3];
  const float* W1a  = (const float*)d_in[4];
  const float* b1a  = (const float*)d_in[5];
  const float* W2a  = (const float*)d_in[6];
  const float* b2a  = (const float*)d_in[7];
  const float* g1   = (const float*)d_in[8];
  const float* be1  = (const float*)d_in[9];
  const float* W1b  = (const float*)d_in[10];
  const float* b1b  = (const float*)d_in[11];
  const float* W2b  = (const float*)d_in[12];
  const float* b2b  = (const float*)d_in[13];
  const float* g2   = (const float*)d_in[14];
  const float* be2  = (const float*)d_in[15];
  float* out = (float*)d_out;

  char* ws = (char*)d_ws;
  size_t off = 0;
  auto carve = [&](size_t bytes) -> void* {
    void* p = ws + off;
    off += (bytes + 255) & ~(size_t)255;
    return p;
  };
  float*  summed = (float*) carve((size_t)NN * HD * sizeof(float));
  float*  counts = (float*) carve((size_t)NN * sizeof(float));
  bf16_t* W1a_h  = (bf16_t*)carve((size_t)(FD + HD) * HD * sizeof(bf16_t));
  bf16_t* W2a_h  = (bf16_t*)carve((size_t)HD * HD * sizeof(bf16_t));
  bf16_t* W1b_h  = (bf16_t*)carve((size_t)(FD + 2 * HD) * HD * sizeof(bf16_t));
  bf16_t* W2b_h  = (bf16_t*)carve((size_t)HD * HD * sizeof(bf16_t));

  zero_f32<<<(NN * HD + 255) / 256, 256, 0, stream>>>(summed, NN * HD);
  zero_f32<<<(NN + 255) / 256, 256, 0, stream>>>(counts, NN);
  cvt_f32_bf16<<<((FD + HD) * HD + 255) / 256, 256, 0, stream>>>(W1a, W1a_h, (FD + HD) * HD);
  cvt_f32_bf16<<<(HD * HD + 255) / 256, 256, 0, stream>>>(W2a, W2a_h, HD * HD);
  cvt_f32_bf16<<<((FD + 2 * HD) * HD + 255) / 256, 256, 0, stream>>>(W1b, W1b_h, (FD + 2 * HD) * HD);
  cvt_f32_bf16<<<(HD * HD + 255) / 256, 256, 0, stream>>>(W2b, W2b_h, HD * HD);

  // 800000 / 32 = 25000 blocks, exact; 50000 / 16 = 3125 blocks, exact.
  edge_kernel<<<NE / 32, 32, 0, stream>>>(x, eidx, eatt, W1a_h, b1a, W2a_h, b2a,
                                          g1, be1, summed, counts);
  node_kernel<<<NN / 16, 32, 0, stream>>>(x, u, summed, counts, W1b_h, b1b,
                                          W2b_h, b2b, g2, be2, out);
}
